// TCA_Block_53343493816497
// MI455X (gfx1250) — matrix operand order repaired and checked by reading, not compiled, not run on hardware
//
#include <hip/hip_runtime.h>
#include <math.h>

typedef __attribute__((ext_vector_type(16))) _Float16 v16h;
typedef __attribute__((ext_vector_type(8)))  _Float16 h8;
typedef __attribute__((ext_vector_type(4)))  _Float16 h4;
typedef __attribute__((ext_vector_type(8)))  float    v8f;
typedef __attribute__((ext_vector_type(4)))  unsigned int v4u;
typedef __attribute__((ext_vector_type(8)))  int      v8i;
typedef __attribute__((ext_vector_type(4)))  int      v4i;

#define DEV static __device__ __forceinline__

constexpr int DIMC  = 768;   // channels
constexpr int SLICE = 12;    // groups
constexpr int CSG   = 64;    // channels per group
constexpr int NHEAD = 16;
constexpr int HDIM  = 48;    // channel-attention size per head
constexpr int NPOS  = 64;    // 8x8 window positions
constexpr int NB    = 2;     // batch
constexpr int GE    = 4096;  // 64ch x 64pos group slice

DEV float gelu_f(float x) { return 0.5f * x * (1.0f + erff(x * 0.70710678118654752f)); }

// ------------- WMMA 16x16x32 f16 fragment loaders (CDNA5 ISA 7.12.2, wave32) --------
// A fragment from row-major [M][K]: per lane = one contiguous 16-half run -> 2x b128.
DEV v16h loadA16(const _Float16* src, int ld, int lane) {
  const int m = lane & 15, h = lane >> 4;
  const _Float16* p = src + m * ld + 16 * h;
  h8 lo = *(const h8*)p;          // K = 16h + 0..7
  h8 hi = *(const h8*)(p + 8);    // K = 16h + 8..15
  return __builtin_shufflevector(lo, hi, 0, 1, 2, 3, 4, 5, 6, 7,
                                 8, 9, 10, 11, 12, 13, 14, 15);
}

// B fragment from a row-major [N][K] source (logical B = srcT): one contiguous 16-half run.
DEV v16h loadB16T(const _Float16* src, int ld, int lane) {
  const int n = lane & 15, h = lane >> 4;
  const _Float16* p = src + n * ld + 16 * h;
  h8 lo = *(const h8*)p;          // K = 16h + 0..7
  h8 hi = *(const h8*)(p + 8);    // K = 16h + 8..15
  return __builtin_shufflevector(lo, hi, 0, 1, 2, 3, 4, 5, 6, 7,
                                 8, 9, 10, 11, 12, 13, 14, 15);
}

// B fragment from row-major [K][N] (column-strided; rows clamped to kmax).
DEV v16h loadB16(const _Float16* src, int ld, int lane, int kmax) {
  v16h b;
  const int n = lane & 15, h = lane >> 4;
#pragma unroll
  for (int e = 0; e < 16; ++e) {
    int k = 16 * h + e;
    if (k >= kmax) k = kmax - 1;
    b[e] = src[k * ld + n];
  }
  return b;
}

// D[M][N] = alpha * A[M][K] @ B + biasLds[m]; compile-time flags select the paths.
template <bool BT, bool HASBIAS, bool F32OUT>
DEV void wmma_gemm(const _Float16* A, int lda,
                   const _Float16* B, int ldb, int kmax,
                   void* Cout, int ldc,
                   const float* biasLds, float alpha,
                   int M, int N, int K, int tid)
{
  const int lane = tid & 31, wv = tid >> 5, nw = (int)(blockDim.x >> 5);
  const int tM = M >> 4, tN = N >> 4;
  for (int t = wv; t < tM * tN; t += nw) {
    const int tm = t / tN, tn = t % tN;
    v8f acc = {};
    for (int k0 = 0; k0 < K; k0 += 32) {
      v16h a = loadA16(A + tm * 16 * lda + k0, lda, lane);
      v16h b;
      if constexpr (BT) b = loadB16T(B + tn * 16 * ldb + k0, ldb, lane);
      else              b = loadB16(B + k0 * ldb + tn * 16, ldb, lane, kmax - k0);
      acc = __builtin_amdgcn_wmma_f32_16x16x32_f16(false, a, false, b, (short)0, acc,
                                                   false, false);
    }
    const int n  = tn * 16 + (lane & 15);
    const int mb = tm * 16 + 8 * (lane >> 4);
    float4 b0, b1;
    if constexpr (HASBIAS) {              // two ds_load_b128 per tile
      b0 = *(const float4*)(biasLds + mb);
      b1 = *(const float4*)(biasLds + mb + 4);
    }
#pragma unroll
    for (int r = 0; r < 8; ++r) {
      float v = acc[r] * alpha;
      if constexpr (HASBIAS)
        v += (r < 4) ? (&b0.x)[r] : (&b1.x)[r - 4];
      if constexpr (F32OUT) ((float*)Cout)[(mb + r) * ldc + n] = v;
      else                  ((_Float16*)Cout)[(mb + r) * ldc + n] = (_Float16)v;
    }
  }
}

// wave32 shuffle reduction + 8-wave LDS combine (r16 = 16 floats of LDS scratch)
DEV void reduce2(float& a, float& b, float* r16, int tid) {
#pragma unroll
  for (int off = 16; off > 0; off >>= 1) {
    a += __shfl_xor(a, off, 32);
    b += __shfl_xor(b, off, 32);
  }
  if ((tid & 31) == 0) { r16[(tid >> 5) * 2] = a; r16[(tid >> 5) * 2 + 1] = b; }
  __syncthreads();
  float ta = 0.f, tb = 0.f;
#pragma unroll
  for (int i = 0; i < 8; ++i) { ta += r16[2 * i]; tb += r16[2 * i + 1]; }
  __syncthreads();
  a = ta; b = tb;
}

// vectorized f32 global -> f16 LDS weight staging (4096 elements)
DEV void stage_w(_Float16* dst, const float* src, int tid) {
  const float4* s4 = (const float4*)src;
  for (int e4 = tid; e4 < GE / 4; e4 += 256) {
    const float4 w = s4[e4];
    h4 o; o[0] = (_Float16)w.x; o[1] = (_Float16)w.y;
    o[2] = (_Float16)w.z; o[3] = (_Float16)w.w;
    *(h4*)(dst + e4 * 4) = o;
  }
}

// ============ K1: fused per-window norm1 + CPE + qkv + channel-attention + proj ======
__global__ __launch_bounds__(256) void tca_window_kernel(
    const float* __restrict__ y,
    const float* __restrict__ n1w, const float* __restrict__ n1b,
    const float* __restrict__ cw,  const float* __restrict__ cb,
    const float* __restrict__ cnw, const float* __restrict__ cnb,
    const float* __restrict__ qw,  const float* __restrict__ qb,
    const float* __restrict__ kw,  const float* __restrict__ kb,
    const float* __restrict__ vw,  const float* __restrict__ vb,
    const float* __restrict__ pw,  const float* __restrict__ pb,
    float* __restrict__ out, float* __restrict__ stats)
{
  extern __shared__ char smem[];
  _Float16* sq = (_Float16*)smem;               // 768x64 f16 [c][n] (q, later attn out)
  _Float16* sk = sq + DIMC * NPOS;              // 768x64 f16 [c][n]
  _Float16* sv = sk + DIMC * NPOS;              // 768x64 f16 [c][n]
  char*  arena = smem + (size_t)3 * DIMC * NPOS * sizeof(_Float16); // 24KB reusable
  float* sBias = (float*)(arena + 24576);       // 64 floats
  float* red   = sBias + 64;                    // 16 floats

  const int tid = threadIdx.x;
  const int wi  = blockIdx.x;
  const int b   = wi >> 10;
  const int wy  = (wi >> 5) & 31;
  const int wx  = wi & 31;

  _Float16* t1T   = (_Float16*)arena;   // yn transposed [n][c] (64x64)
  _Float16* t2    = t1T + GE;           // CPE feat, linear e = c*64+n
  _Float16* wbufA = t2 + GE;            // staged f16 weights [cout][cin]
  const float scale = 0.14433756729740643f; // 48^-0.5

  // ---- per-group: GroupNorm1, ConvPosEnc, q/k/v grouped GEMMs ----
  for (int g = 0; g < SLICE; ++g) {
    if (g + 1 < SLICE) __builtin_prefetch(qw + (g + 1) * GE, 0, 0);
    float s1 = 0.f, s2 = 0.f;
    for (int e = tid; e < GE; e += 256) {
      const int c = e >> 6, n = e & 63;
      const int gi = ((b * DIMC + g * CSG + c) << 16) +
                     ((wy * 8 + (n >> 3)) << 8) + wx * 8 + (n & 7);
      const float v = y[gi];
      s1 += v; s2 += v * v;
    }
    reduce2(s1, s2, red, tid);
    const float mu = s1 * (1.f / GE);
    const float rs = rsqrtf(s2 * (1.f / GE) - mu * mu + 1e-5f);
    for (int e = tid; e < GE; e += 256) {
      const int c = e >> 6, n = e & 63;
      const int cg = g * CSG + c;
      const int gi = ((b * DIMC + cg) << 16) + ((wy * 8 + (n >> 3)) << 8) + wx * 8 + (n & 7);
      t1T[n * 64 + c] = (_Float16)(((y[gi] - mu) * rs) * n1w[cg] + n1b[cg]);
    }
    __syncthreads();
    // depthwise 3x3 inside the 8x8 window (zero padded)
    for (int e = tid; e < GE; e += 256) {
      const int c = e >> 6, n = e & 63;
      const int cg = g * CSG + c;
      const int py = n >> 3, px = n & 7;
      float acc = cb[cg];
      for (int ky = 0; ky < 3; ++ky)
        for (int kx = 0; kx < 3; ++kx) {
          const int iy = py + ky - 1, ix = px + kx - 1;
          if (iy >= 0 && iy < 8 && ix >= 0 && ix < 8)
            acc += (float)t1T[(iy * 8 + ix) * 64 + c] * cw[cg * 9 + ky * 3 + kx];
        }
      t2[e] = (_Float16)acc;
    }
    __syncthreads();
    s1 = 0.f; s2 = 0.f;
    for (int e = tid; e < GE; e += 256) {
      const float v = (float)t2[e];
      s1 += v; s2 += v * v;
    }
    reduce2(s1, s2, red, tid);
    const float mu2 = s1 * (1.f / GE);
    const float rs2 = rsqrtf(s2 * (1.f / GE) - mu2 * mu2 + 1e-5f);
    for (int e = tid; e < GE; e += 256) {
      const int c = e >> 6, n = e & 63;
      const int cg = g * CSG + c;
      const float f = ((float)t2[e] - mu2) * rs2 * cnw[cg] + cnb[cg];
      t1T[n * 64 + c] = (_Float16)((float)t1T[n * 64 + c] + gelu_f(f));
    }
    __syncthreads();

    // grouped 1x1 convs as 64x64x64 WMMA GEMMs (B side transposed -> contiguous loads)
    stage_w(wbufA, qw + g * GE, tid);
    if (tid < 64) sBias[tid] = qb[g * CSG + tid];
    __syncthreads();
    wmma_gemm<true, true, false>(wbufA, 64, t1T, 64, 64, sq + g * GE, 64,
                                 sBias, 1.f, 64, 64, 64, tid);
    __syncthreads();
    stage_w(wbufA, kw + g * GE, tid);
    if (tid < 64) sBias[tid] = kb[g * CSG + tid];
    __syncthreads();
    wmma_gemm<true, true, false>(wbufA, 64, t1T, 64, 64, sk + g * GE, 64,
                                 sBias, 1.f, 64, 64, 64, tid);
    __syncthreads();
    stage_w(wbufA, vw + g * GE, tid);
    if (tid < 64) sBias[tid] = vb[g * CSG + tid];
    __syncthreads();
    wmma_gemm<true, true, false>(wbufA, 64, t1T, 64, 64, sv + g * GE, 64,
                                 sBias, 1.f, 64, 64, 64, tid);
    __syncthreads();
  }

  // ---- channel attention: head h uses channels c = d*16 + h (row stride 16) ----
  float*    attnF = (float*)arena;                                    // 48x48 f32
  _Float16* a16   = (_Float16*)(arena + HDIM * HDIM * sizeof(float)); // 48x64 f16
  for (int h = 0; h < NHEAD; ++h) {
    // attn = scale * Qh(48x64) @ Kh(48x64)^T     (both sides contiguous loads)
    wmma_gemm<true, false, true>(sq + h * 64, 1024, sk + h * 64, 1024, 64,
                                 attnF, HDIM, nullptr, scale, HDIM, HDIM, 64, tid);
    __syncthreads();
    if (tid < HDIM) {
      const int row = tid;
      const int end = ((row >> 2) + 1) << 2;   // block-causal mask over 4-wide blocks
      float mx = -3.0e38f;
      for (int c2 = 0; c2 < end; ++c2) mx = fmaxf(mx, attnF[row * HDIM + c2]);
      float sum = 0.f;
      for (int c2 = 0; c2 < end; ++c2) {
        const float ex = expf(attnF[row * HDIM + c2] - mx);
        attnF[row * HDIM + c2] = ex;
        sum += ex;
      }
      const float inv = 1.f / sum;
      for (int c2 = 0; c2 < 64; ++c2)
        a16[row * 64 + c2] =
            (c2 < end) ? (_Float16)(attnF[row * HDIM + c2] * inv) : (_Float16)0.f;
    }
    __syncthreads();
    // out_h(48x64) = attn(48x48 padded) @ Vh; overwrite q rows of head h
    wmma_gemm<false, false, false>(a16, 64, sv + h * 64, 1024, HDIM,
                                   sq + h * 64, 1024, nullptr, 1.f, HDIM, 64, 64, tid);
    __syncthreads();
  }

  // ---- projection + residual + partial stats for global GroupNorm2 ----
  _Float16* wbufP = (_Float16*)arena;
  float*    Dbuf  = (float*)(arena + GE * sizeof(_Float16));
  for (int g = 0; g < SLICE; ++g) {
    stage_w(wbufP, pw + g * GE, tid);
    __syncthreads();
    wmma_gemm<false, false, true>(wbufP, 64, sq + g * GE, 64, 64,
                                  Dbuf, 64, nullptr, 1.f, 64, 64, 64, tid);
    __syncthreads();
    float s1 = 0.f, s2 = 0.f;
    for (int e = tid; e < GE; e += 256) {
      const int c = e >> 6, n = e & 63;
      const int cg = g * CSG + c;
      const int gi = ((b * DIMC + cg) << 16) + ((wy * 8 + (n >> 3)) << 8) + wx * 8 + (n & 7);
      const float tgt = Dbuf[e] + pb[cg] + y[gi];
      out[gi] = tgt;
      s1 += tgt; s2 += tgt * tgt;
    }
    reduce2(s1, s2, red, tid);
    if (tid == 0) {
      atomicAdd(&stats[(b * SLICE + g) * 2 + 0], s1);
      atomicAdd(&stats[(b * SLICE + g) * 2 + 1], s2);
    }
    __syncthreads();
  }
}

// ============ K0/K2: stats init + finalize (mu, rsig per batch*group) ================
__global__ void init_stats_kernel(float* s) {
  if (threadIdx.x < 2 * NB * SLICE) s[threadIdx.x] = 0.f;
}

__global__ void finalize_stats_kernel(float* s) {
  const int i = threadIdx.x;
  if (i < NB * SLICE) {
    const float Ng = (float)CSG * 65536.f;
    const float mu = s[2 * i] / Ng;
    const float var = s[2 * i + 1] / Ng - mu * mu;
    s[2 * i]     = mu;
    s[2 * i + 1] = rsqrtf(var + 1e-5f);
  }
}

// ============ K3: GroupNorm2 + grouped MLP (gconv -> GELU -> gconv) + residual =======
__global__ __launch_bounds__(256) void tca_mlp_kernel(
    float* __restrict__ out,
    const float* __restrict__ n2w, const float* __restrict__ n2b,
    const float* __restrict__ f1w, const float* __restrict__ f1b,
    const float* __restrict__ f2w, const float* __restrict__ f2b,
    const float* __restrict__ stats)
{
  __shared__ __attribute__((aligned(16))) float    tgtS[GE];  // [c][n]
  __shared__ __attribute__((aligned(16))) _Float16 xnT[GE];   // [n][c]
  __shared__ __attribute__((aligned(16))) _Float16 wb[GE];
  __shared__ __attribute__((aligned(16))) _Float16 hbT[GE];   // [n][c]
  __shared__ __attribute__((aligned(16))) float    Db[GE];    // [c][n]
  __shared__ __attribute__((aligned(16))) float sb1[CSG], sb2[CSG];
  const int tid   = threadIdx.x;
  const int g     = blockIdx.x % SLICE;
  const int chunk = (blockIdx.x / SLICE) & 1023;  // 64-position slab in the HW plane
  const int b     = blockIdx.x / (SLICE * 1024);
  const float mu = stats[(b * SLICE + g) * 2 + 0];
  const float rs = stats[(b * SLICE + g) * 2 + 1];
  const int base = ((b * DIMC + g * CSG) << 16) + chunk * 64; // element 0 of slab

  // ---- stage the 64x64 f32 tgt slab into LDS ----
#if __has_builtin(__builtin_amdgcn_tensor_load_to_lds)
  if (tid < 32) {
    // Tensor DMA descriptor (CDNA5 ISA ch.8): 2D tile 64x64 f32, dim0 stride = H*W.
    const unsigned long long ga = (unsigned long long)(const void*)(out + base);
    const unsigned lds = (unsigned)(unsigned long long)(const void*)tgtS; // addr[31:0] = LDS offset
    v4u g0;
    g0[0] = 1u;                                     // count=1, user descriptor
    g0[1] = lds;                                    // lds_addr
    g0[2] = (unsigned)(ga & 0xffffffffu);           // global_addr[31:0]
    g0[3] = (unsigned)((ga >> 32) & 0x01ffffffu) |  // global_addr[56:32]
            (2u << 30);                             // type = 2 (image)
    v8i g1;
    g1[0] = (int)(2u << 16);        // data_size = 4B
    g1[1] = (int)(64u << 16);       // tensor_dim0[15:0] = 64  (bits 63:48)
    g1[2] = (int)(64u << 16);       // tensor_dim1[15:0] = 64  (bits 95:80)
    g1[3] = (int)(64u << 16);       // tile_dim0 = 64          (bits 127:112)
    g1[4] = 64;                     // tile_dim1 = 64          (bits 143:128)
    g1[5] = 65536;                  // tensor_dim0_stride low32 (bits 191:160)
    g1[6] = 0;
    g1[7] = 0;
    v4i g2 = {0, 0, 0, 0}, g3 = {0, 0, 0, 0};
    v8i g4 = {0, 0, 0, 0, 0, 0, 0, 0};
    __builtin_amdgcn_tensor_load_to_lds(g0, g1, g2, g3, g4, 0);
  }
  // stage weights/bias while the TDM engine streams the tile
  stage_w(wb, f1w + g * GE, tid);
  if (tid < 64) { sb1[tid] = f1b[g * CSG + tid]; sb2[tid] = f2b[g * CSG + tid]; }
  if (tid < 32) __builtin_amdgcn_s_wait_tensorcnt(0);
  __syncthreads();
#else
  for (int e = tid; e < GE; e += 256) {
    const int c = e >> 6, n = e & 63;
    tgtS[e] = out[base + (c << 16) + n];
  }
  stage_w(wb, f1w + g * GE, tid);
  if (tid < 64) { sb1[tid] = f1b[g * CSG + tid]; sb2[tid] = f2b[g * CSG + tid]; }
  __syncthreads();
#endif

  for (int e = tid; e < GE; e += 256) {
    const int c = e >> 6, n = e & 63;
    const int cg = g * CSG + c;
    xnT[n * 64 + c] = (_Float16)(((tgtS[e] - mu) * rs) * n2w[cg] + n2b[cg]);
  }
  __syncthreads();
  wmma_gemm<true, true, true>(wb, 64, xnT, 64, 64, Db, 64, sb1, 1.f, 64, 64, 64, tid);
  __syncthreads();
  for (int e = tid; e < GE; e += 256) {
    const int c = e >> 6, n = e & 63;
    hbT[n * 64 + c] = (_Float16)gelu_f(Db[e]);
  }
  stage_w(wb, f2w + g * GE, tid);
  __syncthreads();
  wmma_gemm<true, true, true>(wb, 64, hbT, 64, 64, Db, 64, sb2, 1.f, 64, 64, 64, tid);
  __syncthreads();
  for (int e = tid; e < GE; e += 256) {
    const int c = e >> 6, n = e & 63;
    out[base + (c << 16) + n] = tgtS[e] + Db[e];
  }
}

// =====================================================================================
extern "C" void kernel_launch(void* const* d_in, const int* in_sizes, int n_in,
                              void* d_out, int out_size, void* d_ws, size_t ws_size,
                              hipStream_t stream) {
  (void)in_sizes; (void)n_in; (void)out_size; (void)ws_size;
  const float* y   = (const float*)d_in[0];
  const float* n1w = (const float*)d_in[1];
  const float* n1b = (const float*)d_in[2];
  const float* cw  = (const float*)d_in[3];
  const float* cb  = (const float*)d_in[4];
  const float* cnw = (const float*)d_in[5];
  const float* cnb = (const float*)d_in[6];
  const float* qw  = (const float*)d_in[7];
  const float* qb  = (const float*)d_in[8];
  const float* kw  = (const float*)d_in[9];
  const float* kb  = (const float*)d_in[10];
  const float* vw  = (const float*)d_in[11];
  const float* vb  = (const float*)d_in[12];
  const float* pw  = (const float*)d_in[13];
  const float* pb  = (const float*)d_in[14];
  const float* n2w = (const float*)d_in[15];
  const float* n2b = (const float*)d_in[16];
  const float* f1w = (const float*)d_in[17];
  const float* f1b = (const float*)d_in[18];
  const float* f2w = (const float*)d_in[19];
  const float* f2b = (const float*)d_in[20];
  float* out   = (float*)d_out;
  float* stats = (float*)d_ws;

  // 3x96KB q/k/v + 24KB arena + bias/reduce scratch = 319,808 B (< 320KB WGP LDS)
  constexpr size_t K1_SMEM =
      (size_t)3 * DIMC * NPOS * sizeof(_Float16) + 24576 + 64 * 4 + 16 * 4;
  hipError_t err = hipFuncSetAttribute(
      reinterpret_cast<const void*>(tca_window_kernel),
      hipFuncAttributeMaxDynamicSharedMemorySize, (int)K1_SMEM);
  (void)err;

  init_stats_kernel<<<1, 64, 0, stream>>>(stats);
  tca_window_kernel<<<2048, 256, K1_SMEM, stream>>>(
      y, n1w, n1b, cw, cb, cnw, cnb, qw, qb, kw, kb, vw, vb, pw, pb, out, stats);
  finalize_stats_kernel<<<1, 32, 0, stream>>>(stats);
  tca_mlp_kernel<<<NB * 1024 * SLICE, 256, 0, stream>>>(
      out, n2w, n2b, f1w, f1b, f2w, f2b, stats);
}
